// Kernel_90340342104308
// MI455X (gfx1250) — compile-verified, matrix operand order checked by reading, not run on hardware
//
#include <hip/hip_runtime.h>
#include <hip/hip_bf16.h>

typedef __bf16 bf16_t;
typedef __attribute__((ext_vector_type(16))) __bf16 v16bf;
typedef __attribute__((ext_vector_type(8)))  __bf16 v8bf;
typedef __attribute__((ext_vector_type(8)))  float  v8f;
typedef __attribute__((ext_vector_type(4)))  int    v4i;

#define NPTS   8192
#define MPTS   8192
#define IN_DIM 64
#define HID    128
#define TOTPTS (NPTS + MPTS)

// ---------------------------------------------------------------------------
// Async global->LDS staging (gfx1250 GLOBAL_LOAD_ASYNC_TO_LDS_B128, ASYNCcnt).
// Builtin signature (probe-verified): (global v4i*, shared v4i*, imm, imm).
// ---------------------------------------------------------------------------
#if defined(__gfx1250__) && __has_builtin(__builtin_amdgcn_global_load_async_to_lds_b128)
#define ASYNC_LDS_COPY 1
#else
#define ASYNC_LDS_COPY 0
#endif

typedef __attribute__((address_space(1))) v4i g_v4i;   // global-AS pointee
typedef __attribute__((address_space(3))) v4i l_v4i;   // LDS-AS pointee

__device__ inline void async_copy16(void* lds_dst, const void* gsrc) {
#if ASYNC_LDS_COPY
    g_v4i* gp = (g_v4i*)(unsigned long long)gsrc;
    l_v4i* lp = (l_v4i*)(unsigned int)(unsigned long long)lds_dst;
    __builtin_amdgcn_global_load_async_to_lds_b128(gp, lp, 0, 0);
#else
    *(v8bf*)lds_dst = *(const v8bf*)gsrc;
#endif
}

__device__ inline void wait_async() {
#if ASYNC_LDS_COPY
#if __has_builtin(__builtin_amdgcn_s_wait_asynccnt)
    __builtin_amdgcn_s_wait_asynccnt(0);
#else
    asm volatile("s_wait_asynccnt 0" ::: "memory");
#endif
#endif
}

// Raw v_exp_f32 (argument <= 0 here; denorm flush-to-zero is acceptable).
__device__ inline float fast_exp2(float v) {
#if __has_builtin(__builtin_amdgcn_exp2f)
    return __builtin_amdgcn_exp2f(v);
#else
    return exp2f(v);
#endif
}

// ---------------------------------------------------------------------------
// Fragment load for CDNA5 WMMA 16x16x32 bf16 (wave32).
// A (16x32, MxK): lane l holds row m=(l&15); K chunk kb=8*(l>>4):
//   VGPR0..3 = K kb..kb+7, VGPR4..7 = K kb+16..kb+23 (2 bf16 per VGPR).
// B column n == row n of an N-major [n][k] tile -> same per-lane pattern.
// ---------------------------------------------------------------------------
__device__ inline v16bf load_frag(const bf16_t* base, int ld, int row0, int ko, int lane) {
    const int r  = row0 + (lane & 15);
    const int kb = ko + ((lane >> 4) << 3);
    const bf16_t* p = base + r * ld + kb;
    v8bf lo = *(const v8bf*)(p);
    v8bf hi = *(const v8bf*)(p + 16);
    v16bf a;
#pragma unroll
    for (int i = 0; i < 8; ++i) { a[i] = lo[i]; a[i + 8] = hi[i]; }
    return a;
}

__device__ inline float softplus_f(float v) {
    return fmaxf(v, 0.0f) + __logf(1.0f + __expf(-fabsf(v)));
}

// ---------------------------------------------------------------------------
// Kernel 1: fused 3-layer MLP (+skip) for concat(x, z); writes bf16 features
// and fp32 row squared-norms. 128 points per block, 256 threads (8 waves).
// ---------------------------------------------------------------------------
__global__ __launch_bounds__(256) void mlp_kernel(
    const float* __restrict__ x,  const float* __restrict__ z,
    const float* __restrict__ w1, const float* __restrict__ b1,
    const float* __restrict__ w2, const float* __restrict__ b2,
    const float* __restrict__ w3, const float* __restrict__ b3,
    const float* __restrict__ skw,
    bf16_t* __restrict__ f_out, float* __restrict__ sq_out)
{
    __shared__ __align__(16) bf16_t Xt[128 * IN_DIM];
    __shared__ __align__(16) bf16_t Wb[HID * HID];      // W2 then W3
    __shared__ __align__(16) bf16_t Ws[HID * IN_DIM];   // W1 then skip
    __shared__ __align__(16) bf16_t H1[128 * HID];
    __shared__ __align__(16) bf16_t H2[128 * HID];
    __shared__ float bias[HID];
    __shared__ float sq[128];

    const int tid  = threadIdx.x;
    const int lane = tid & 31;
    const int wv   = tid >> 5;
    const int p0   = blockIdx.x * 128;
    const int m0   = wv * 16;

    for (int i = tid; i < 128 * IN_DIM; i += 256) {
        const int r = i >> 6, c = i & 63;
        const int p = p0 + r;
        const float v = (p < NPTS) ? x[(size_t)p * IN_DIM + c]
                                   : z[(size_t)(p - NPTS) * IN_DIM + c];
        Xt[r * IN_DIM + c] = (bf16_t)v;
    }
    for (int i = tid; i < HID * IN_DIM; i += 256) Ws[i] = (bf16_t)w1[i];
    if (tid < HID) bias[tid] = b1[tid];
    __syncthreads();

    // ---- Layer 1: H1 = softplus(X @ W1.T + b1), K = 64 ----
    for (int n0 = 0; n0 < HID; n0 += 16) {
        v8f c = {};
#pragma unroll
        for (int ko = 0; ko < IN_DIM; ko += 32) {
            v16bf a = load_frag(Xt, IN_DIM, m0, ko, lane);
            v16bf b = load_frag(Ws, IN_DIM, n0, ko, lane);
            c = __builtin_amdgcn_wmma_f32_16x16x32_bf16(false, a, false, b, (short)0, c, false, false);
        }
        const int n = n0 + (lane & 15);
        const float bn = bias[n];
#pragma unroll
        for (int v = 0; v < 8; ++v) {
            const int m = m0 + v + ((lane >> 4) << 3);
            H1[m * HID + n] = (bf16_t)softplus_f(c[v] + bn);
        }
    }
    __syncthreads();

    for (int i = tid; i < HID * HID; i += 256) Wb[i] = (bf16_t)w2[i];
    if (tid < HID) bias[tid] = b2[tid];
    __syncthreads();

    // ---- Layer 2: H2 = softplus(H1 @ W2.T + b2), K = 128 ----
    for (int n0 = 0; n0 < HID; n0 += 16) {
        v8f c = {};
#pragma unroll
        for (int ko = 0; ko < HID; ko += 32) {
            v16bf a = load_frag(H1, HID, m0, ko, lane);
            v16bf b = load_frag(Wb, HID, n0, ko, lane);
            c = __builtin_amdgcn_wmma_f32_16x16x32_bf16(false, a, false, b, (short)0, c, false, false);
        }
        const int n = n0 + (lane & 15);
        const float bn = bias[n];
#pragma unroll
        for (int v = 0; v < 8; ++v) {
            const int m = m0 + v + ((lane >> 4) << 3);
            H2[m * HID + n] = (bf16_t)softplus_f(c[v] + bn);
        }
    }
    __syncthreads();

    for (int i = tid; i < HID * HID; i += 256) Wb[i] = (bf16_t)w3[i];
    for (int i = tid; i < HID * IN_DIM; i += 256) Ws[i] = (bf16_t)skw[i];
    if (tid < HID) bias[tid] = b3[tid];
    if (tid < 128) sq[tid] = 0.0f;
    __syncthreads();

    // ---- Layer 3: F = H2 @ W3.T + b3 + X @ skip.T ----
    float s2[8];
#pragma unroll
    for (int v = 0; v < 8; ++v) s2[v] = 0.0f;

    for (int n0 = 0; n0 < HID; n0 += 16) {
        v8f c = {};
#pragma unroll
        for (int ko = 0; ko < HID; ko += 32) {
            v16bf a = load_frag(H2, HID, m0, ko, lane);
            v16bf b = load_frag(Wb, HID, n0, ko, lane);
            c = __builtin_amdgcn_wmma_f32_16x16x32_bf16(false, a, false, b, (short)0, c, false, false);
        }
#pragma unroll
        for (int ko = 0; ko < IN_DIM; ko += 32) {
            v16bf a = load_frag(Xt, IN_DIM, m0, ko, lane);
            v16bf b = load_frag(Ws, IN_DIM, n0, ko, lane);
            c = __builtin_amdgcn_wmma_f32_16x16x32_bf16(false, a, false, b, (short)0, c, false, false);
        }
        const int n = n0 + (lane & 15);
        const float bn = bias[n];
#pragma unroll
        for (int v = 0; v < 8; ++v) {
            const int m = m0 + v + ((lane >> 4) << 3);
            const bf16_t fb = (bf16_t)(c[v] + bn);
            const float fv = (float)fb;
            f_out[(size_t)(p0 + m) * HID + n] = fb;
            s2[v] += fv * fv;
        }
    }
#pragma unroll
    for (int v = 0; v < 8; ++v) {
        const int m = m0 + v + ((lane >> 4) << 3);
        atomicAdd(&sq[m], s2[v]);
    }
    __syncthreads();
    if (tid < 128) sq_out[p0 + tid] = sq[tid];
}

// ---------------------------------------------------------------------------
// Kernel 2: pairwise RBF. 256x128 output tile per block, 256 threads.
// Each wave owns two 16-row blocks sharing B fragments (8 WMMAs / n0 step).
// ---------------------------------------------------------------------------
__global__ __launch_bounds__(256) void pairwise_kernel(
    const bf16_t* __restrict__ f, const float* __restrict__ sqall,
    const float* __restrict__ log_sigma, float* __restrict__ out)
{
    __shared__ __align__(16) bf16_t At[256 * HID];   // 64 KB
    __shared__ __align__(16) bf16_t Bt[128 * HID];   // 32 KB
    __shared__ float sqa[256];
    __shared__ float sqb[128];

    const int tid  = threadIdx.x;
    const int lane = tid & 31;
    const int wv   = tid >> 5;
    const int i0   = blockIdx.y * 256;
    const int j0   = blockIdx.x * 128;
    const bf16_t* fx = f;
    const bf16_t* fz = f + (size_t)NPTS * HID;

    // Stage tiles: async global->LDS (b128 granules, ASYNCcnt-tracked).
    for (int i = tid * 8; i < 256 * HID; i += 256 * 8)
        async_copy16(&At[i], &fx[(size_t)i0 * HID + i]);
    for (int i = tid * 8; i < 128 * HID; i += 256 * 8)
        async_copy16(&Bt[i], &fz[(size_t)j0 * HID + i]);
    sqa[tid] = sqall[i0 + tid];
    if (tid < 128) sqb[tid] = sqall[NPTS + j0 + tid];
    wait_async();
    __syncthreads();

    // Fold log2(e) into the bandwidth factor: exp(a*d) == exp2(a*log2e*d).
    const float inv_bw2 = -0.5f * 1.442695041f * __powf(10.0f, -log_sigma[0]);
    const int m0a = wv * 32;
    const int m0b = m0a + 16;

    // Hoist A fragments for both row blocks, all 4 K-steps.
    v16bf afragA[4], afragB[4];
#pragma unroll
    for (int kk = 0; kk < 4; ++kk) {
        afragA[kk] = load_frag(At, HID, m0a, kk * 32, lane);
        afragB[kk] = load_frag(At, HID, m0b, kk * 32, lane);
    }
    float sqmA[8], sqmB[8];
#pragma unroll
    for (int v = 0; v < 8; ++v) {
        const int dm = v + ((lane >> 4) << 3);
        sqmA[v] = sqa[m0a + dm];
        sqmB[v] = sqa[m0b + dm];
    }

#pragma unroll 4
    for (int n0 = 0; n0 < 128; n0 += 16) {
        v8f ca = {}, cb = {};
#pragma unroll
        for (int kk = 0; kk < 4; ++kk) {
            v16bf b = load_frag(Bt, HID, n0, kk * 32, lane);
            ca = __builtin_amdgcn_wmma_f32_16x16x32_bf16(false, afragA[kk], false, b, (short)0, ca, false, false);
            cb = __builtin_amdgcn_wmma_f32_16x16x32_bf16(false, afragB[kk], false, b, (short)0, cb, false, false);
        }
        const int n = n0 + (lane & 15);
        const float sn = sqb[n];
        const size_t col = (size_t)(j0 + n);
#pragma unroll
        for (int v = 0; v < 8; ++v) {
            const int dm = v + ((lane >> 4) << 3);
            const float da = sqmA[v] + sn - 2.0f * ca[v];
            const float db = sqmB[v] + sn - 2.0f * cb[v];
            __builtin_nontemporal_store(fast_exp2(inv_bw2 * da),
                                        &out[(size_t)(i0 + m0a + dm) * MPTS + col]);
            __builtin_nontemporal_store(fast_exp2(inv_bw2 * db),
                                        &out[(size_t)(i0 + m0b + dm) * MPTS + col]);
        }
    }
}

// ---------------------------------------------------------------------------
extern "C" void kernel_launch(void* const* d_in, const int* in_sizes, int n_in,
                              void* d_out, int out_size, void* d_ws, size_t ws_size,
                              hipStream_t stream) {
    (void)in_sizes; (void)n_in; (void)out_size; (void)ws_size;
    const float* x    = (const float*)d_in[0];
    const float* z    = (const float*)d_in[1];
    const float* w1   = (const float*)d_in[2];
    const float* b1   = (const float*)d_in[3];
    const float* w2   = (const float*)d_in[4];
    const float* b2   = (const float*)d_in[5];
    const float* w3   = (const float*)d_in[6];
    const float* b3   = (const float*)d_in[7];
    const float* skw  = (const float*)d_in[8];
    const float* lsig = (const float*)d_in[9];

    bf16_t* f  = (bf16_t*)d_ws;                                               // [16384][128] bf16
    float*  sq = (float*)((char*)d_ws + (size_t)TOTPTS * HID * sizeof(bf16_t)); // [16384] f32

    mlp_kernel<<<TOTPTS / 128, 256, 0, stream>>>(x, z, w1, b1, w2, b2, w3, b3, skw, f, sq);

    dim3 grid(MPTS / 128, NPTS / 256);
    pairwise_kernel<<<grid, 256, 0, stream>>>(f, sq, lsig, (float*)d_out);
}
